// ImprovedGNNClassifier_7533372637467
// MI455X (gfx1250) — matrix operand order repaired and checked by reading, not compiled, not run on hardware
//
#include <hip/hip_runtime.h>
#include <stdint.h>

// ---------------------------------------------------------------------------
// 2-layer GCN (GCNConv -> BN -> ReLU -> GCNConv -> log_softmax) for gfx1250.
// GEMMs: v_wmma_f32_16x16x32_bf16 with double-buffered async global->LDS
// staging (ASYNCcnt). Edge aggregation: f32 HW atomics (global_atomic_add_f32).
// ---------------------------------------------------------------------------

typedef __attribute__((ext_vector_type(16))) __bf16 v16bf;
typedef __attribute__((ext_vector_type(8)))  float  v8f;
typedef __attribute__((ext_vector_type(4)))  int    v4i;

// AS-qualified element types for the async-copy builtin's pointer params.
typedef __attribute__((address_space(1))) v4i gv4i;   // global
typedef __attribute__((address_space(3))) v4i lv4i;   // LDS

union FragBF { uint4 q[2]; v16bf v; };

#if defined(__has_builtin) && __has_builtin(__builtin_amdgcn_global_load_async_to_lds_b128)
#define USE_ASYNC_LDS 1
#if __has_builtin(__builtin_amdgcn_s_wait_asynccnt)
#define WAIT_ASYNC() __builtin_amdgcn_s_wait_asynccnt(0)
#else
#define WAIT_ASYNC() asm volatile("s_wait_asynccnt 0x0" ::: "memory")
#endif
#else
#define USE_ASYNC_LDS 0
#define WAIT_ASYNC() ((void)0)
#endif

#if USE_ASYNC_LDS
static __device__ __forceinline__ void asyncCopyB128(const void* g, void* l) {
  __builtin_amdgcn_global_load_async_to_lds_b128((gv4i*)g, (lv4i*)l, 0, 0);
}
#else
static __device__ __forceinline__ void asyncCopyB128(const void* g, void* l) {
  *(uint4*)l = *(const uint4*)g;
}
#endif

static __device__ __forceinline__ unsigned short f2bf(float f) {
  unsigned u = __builtin_bit_cast(unsigned, f);
  u += 0x7FFFu + ((u >> 16) & 1u);       // round-to-nearest-even
  return (unsigned short)(u >> 16);
}

static __device__ __forceinline__ void atomAddF(float* p, float v) {
  unsafeAtomicAdd(p, v);                  // global_atomic_add_f32
}

// ---------------- casts / transposes ---------------------------------------

__global__ void k_cast_bf16(const float* __restrict__ src,
                            unsigned short* __restrict__ dst, int n) {
  int i = blockIdx.x * 256 + threadIdx.x;
  if (i < n) dst[i] = f2bf(src[i]);
}

// w1t[n][k] = bf16(W1[k][n]),  W1 is [256][256]
__global__ void k_w1_tcast(const float* __restrict__ W1,
                           unsigned short* __restrict__ w1t) {
  int idx = blockIdx.x * 256 + threadIdx.x;     // 65536 total
  int n = idx >> 8, k = idx & 255;
  w1t[idx] = f2bf(W1[k * 256 + n]);
}

// w2t[n][k] = bf16(W2[k][n]) for n<40 else 0,  W2 is [256][40], w2t is [64][256]
__global__ void k_w2_tcast(const float* __restrict__ W2,
                           unsigned short* __restrict__ w2t) {
  int idx = blockIdx.x * 256 + threadIdx.x;     // 16384 total
  int n = idx >> 8, k = idx & 255;
  float v = (n < 40) ? W2[k * 40 + n] : 0.0f;
  w2t[idx] = f2bf(v);
}

// ---------------- degree / norm --------------------------------------------

__global__ void k_set_deg(float* __restrict__ deg, int n) {
  int i = blockIdx.x * 256 + threadIdx.x;
  if (i < n) deg[i] = 1.0f;                     // self-loop
}

__global__ void k_acc_deg(const int* __restrict__ ei, float* __restrict__ deg, int E) {
  int e = blockIdx.x * 256 + threadIdx.x;
  if (e < E) atomAddF(&deg[ei[E + e]], 1.0f);   // dst = ei[1][e]
}

__global__ void k_dinv(const float* __restrict__ deg, float* __restrict__ dinv, int n) {
  int i = blockIdx.x * 256 + threadIdx.x;
  if (i < n) { float d = deg[i]; dinv[i] = (d > 0.0f) ? rsqrtf(d) : 0.0f; }
}

// ---------------- WMMA GEMM: C[M][ldC] = A[M][256](bf16) x B(256 x Ntile) ---
// Bt is B transposed: Bt[n][k], n-major, 256 K per row (bf16).
// Block = 256 threads = 8 waves. Block tile 128(M) x 64(N); wave tile 16 x 64.
// Double-buffered LDS tiles staged with async global->LDS DMA.

__global__ __launch_bounds__(256)
void k_gemm_bf16(const unsigned short* __restrict__ A,
                 const unsigned short* __restrict__ Bt,
                 float* __restrict__ C, int M, int ldC, int storeCols) {
  __shared__ unsigned short As[2][128][32];   // 2 x 8 KB
  __shared__ unsigned short Bs[2][64][32];    // 2 x 4 KB

  const int tid   = threadIdx.x;
  const int wave  = tid >> 5;
  const int lane  = tid & 31;
  const int mBase = blockIdx.y * 128;
  const int nBase = blockIdx.x * 64;

  v8f acc0 = {}, acc1 = {}, acc2 = {}, acc3 = {};

  auto stage = [&](int buf, int k0) {
    // --- A tile: 128 x 32 bf16 = 512 b128 transfers (2 per thread) ---
    #pragma unroll
    for (int half = 0; half < 2; ++half) {
      int i = tid + half * 256;
      int row = i >> 2, ch = i & 3;
      int gr = mBase + row;
      unsigned short* ldst = &As[buf][row][ch * 8];
      if (gr < M) {
        asyncCopyB128(A + (size_t)gr * 256 + k0 + ch * 8, ldst);
      } else {
        *(uint4*)ldst = make_uint4(0u, 0u, 0u, 0u);   // zero-pad edge tile
      }
    }
    // --- Bt tile: 64 x 32 bf16 = 256 b128 transfers (1 per thread) ---
    {
      int row = tid >> 2, ch = tid & 3;
      asyncCopyB128(Bt + (size_t)(nBase + row) * 256 + k0 + ch * 8,
                    &Bs[buf][row][ch * 8]);
    }
  };

  stage(0, 0);
  WAIT_ASYNC();
  __syncthreads();

  for (int kt = 0; kt < 8; ++kt) {            // K = 256, 32 per step
    const int cur = kt & 1;
    if (kt < 7) stage(cur ^ 1, (kt + 1) * 32);   // overlap DMA with WMMA

    // A fragment (16x32): lane<16 -> K0..15, lane>=16 -> K16..31
    FragBF fa;
    {
      int m  = (wave << 4) + (lane & 15);
      int kh = (lane >> 4) * 16;
      fa.q[0] = *(const uint4*)&As[cur][m][kh];
      fa.q[1] = *(const uint4*)&As[cur][m][kh + 8];
    }
    // B fragments (32x16 each): lane<16 -> K0..15 of col N=lane; lane>=16 -> K16..31
    FragBF fb0, fb1, fb2, fb3;
    {
      int n    = lane & 15;
      int koff = (lane >> 4) * 16;
      fb0.q[0] = *(const uint4*)&Bs[cur][n +  0][koff];
      fb0.q[1] = *(const uint4*)&Bs[cur][n +  0][koff + 8];
      fb1.q[0] = *(const uint4*)&Bs[cur][n + 16][koff];
      fb1.q[1] = *(const uint4*)&Bs[cur][n + 16][koff + 8];
      fb2.q[0] = *(const uint4*)&Bs[cur][n + 32][koff];
      fb2.q[1] = *(const uint4*)&Bs[cur][n + 32][koff + 8];
      fb3.q[0] = *(const uint4*)&Bs[cur][n + 48][koff];
      fb3.q[1] = *(const uint4*)&Bs[cur][n + 48][koff + 8];
    }
    acc0 = __builtin_amdgcn_wmma_f32_16x16x32_bf16(false, fa.v, false, fb0.v,
                                                   (short)0, acc0, false, false);
    acc1 = __builtin_amdgcn_wmma_f32_16x16x32_bf16(false, fa.v, false, fb1.v,
                                                   (short)0, acc1, false, false);
    acc2 = __builtin_amdgcn_wmma_f32_16x16x32_bf16(false, fa.v, false, fb2.v,
                                                   (short)0, acc2, false, false);
    acc3 = __builtin_amdgcn_wmma_f32_16x16x32_bf16(false, fa.v, false, fb3.v,
                                                   (short)0, acc3, false, false);

    WAIT_ASYNC();           // next tile's DMA done before anyone reads it
    __syncthreads();
  }

  // D layout: VGPR r -> row = r + 8*(lane>=16), col = lane&15
  int colLocal = lane & 15;
  int rowOff   = (lane >> 4) * 8;
  int mW       = mBase + (wave << 4);
  #pragma unroll
  for (int r = 0; r < 8; ++r) {
    int gr = mW + r + rowOff;
    if (gr >= M) continue;
    float* Crow = C + (size_t)gr * ldC;
    int c0 = nBase + colLocal;
    if (c0      < storeCols) Crow[c0]      = acc0[r];
    if (c0 + 16 < storeCols) Crow[c0 + 16] = acc1[r];
    if (c0 + 32 < storeCols) Crow[c0 + 32] = acc2[r];
    if (c0 + 48 < storeCols) Crow[c0 + 48] = acc3[r];
  }
}

// ---------------- layer-1 aggregation --------------------------------------

// hagg[i][f] = dinv[i]^2 * h0[i][f] + b1[f]   (self-loop + bias)
__global__ void k_init_h(const float* __restrict__ h0, const float* __restrict__ dinv,
                         const float* __restrict__ b1, float* __restrict__ hagg, int total) {
  int idx = blockIdx.x * 256 + threadIdx.x;
  if (idx >= total) return;
  int i = idx >> 8, f = idx & 255;
  float di = dinv[i];
  hagg[idx] = di * di * h0[idx] + b1[f];
}

// one wave per edge: hagg[dst] += w * h0[src] over 256 features (float4)
__global__ __launch_bounds__(256)
void k_edge_agg_h(const int* __restrict__ ei, const float* __restrict__ h0,
                  const float* __restrict__ dinv, float* __restrict__ hagg, int E) {
  int wave = threadIdx.x >> 5, lane = threadIdx.x & 31;
  int e = blockIdx.x * 8 + wave;
  if (e >= E) return;
  int s = ei[e], d = ei[E + e];
  float w = dinv[s] * dinv[d];
  const float4* srow = (const float4*)(h0 + (size_t)s * 256);
  float* drow = hagg + (size_t)d * 256;
  float4 v0 = srow[lane];
  float4 v1 = srow[lane + 32];
  int f0 = lane * 4;
  atomAddF(drow + f0 + 0, w * v0.x);
  atomAddF(drow + f0 + 1, w * v0.y);
  atomAddF(drow + f0 + 2, w * v0.z);
  atomAddF(drow + f0 + 3, w * v0.w);
  atomAddF(drow + 128 + f0 + 0, w * v1.x);
  atomAddF(drow + 128 + f0 + 1, w * v1.y);
  atomAddF(drow + 128 + f0 + 2, w * v1.z);
  atomAddF(drow + 128 + f0 + 3, w * v1.w);
}

// ---------------- batch norm -----------------------------------------------

__global__ void k_bn_zero(float* __restrict__ bnsum, float* __restrict__ bnsq) {
  int c = threadIdx.x;
  bnsum[c] = 0.0f; bnsq[c] = 0.0f;
}

__global__ __launch_bounds__(256)
void k_bn_partial(const float* __restrict__ h, float* __restrict__ bnsum,
                  float* __restrict__ bnsq, int n) {
  int c = threadIdx.x;                       // one column per thread
  float s = 0.0f, s2 = 0.0f;
  for (int r = blockIdx.x; r < n; r += gridDim.x) {
    float v = h[(size_t)r * 256 + c];
    s += v; s2 += v * v;
  }
  atomAddF(&bnsum[c], s);
  atomAddF(&bnsq[c], s2);
}

__global__ void k_bn_final(const float* __restrict__ bnsum, const float* __restrict__ bnsq,
                           const float* __restrict__ gamma, const float* __restrict__ beta,
                           float* __restrict__ scale, float* __restrict__ shift, int n) {
  int c = threadIdx.x;
  float inv = 1.0f / (float)n;
  float mu  = bnsum[c] * inv;
  float var = bnsq[c] * inv - mu * mu;
  float rs  = rsqrtf(var + 1e-5f);
  float sc  = gamma[c] * rs;
  scale[c] = sc;
  shift[c] = beta[c] - mu * sc;
}

__global__ void k_bn_relu_cast(const float* __restrict__ h, const float* __restrict__ scale,
                               const float* __restrict__ shift,
                               unsigned short* __restrict__ hbf, int total) {
  int idx = blockIdx.x * 256 + threadIdx.x;
  if (idx >= total) return;
  int c = idx & 255;
  float v = fmaf(h[idx], scale[c], shift[c]);
  hbf[idx] = f2bf(fmaxf(v, 0.0f));
}

// ---------------- layer-2 aggregation + log_softmax ------------------------

__global__ void k_init_out(const float* __restrict__ z, const float* __restrict__ dinv,
                           const float* __restrict__ b2, float* __restrict__ out, int total) {
  int idx = blockIdx.x * 256 + threadIdx.x;
  if (idx >= total) return;
  int i = idx / 40, c = idx - i * 40;
  float di = dinv[i];
  out[idx] = di * di * z[idx] + b2[c];
}

__global__ __launch_bounds__(256)
void k_edge_agg_out(const int* __restrict__ ei, const float* __restrict__ z,
                    const float* __restrict__ dinv, float* __restrict__ out, int E) {
  int wave = threadIdx.x >> 5, lane = threadIdx.x & 31;
  int e = blockIdx.x * 8 + wave;
  if (e >= E) return;
  int s = ei[e], d = ei[E + e];
  float w = dinv[s] * dinv[d];
  const float* zs = z + (size_t)s * 40;
  float* od = out + (size_t)d * 40;
  atomAddF(od + lane, w * zs[lane]);
  if (lane < 8) atomAddF(od + 32 + lane, w * zs[32 + lane]);
}

__global__ void k_logsoftmax(float* __restrict__ out, int n) {
  int i = blockIdx.x * 256 + threadIdx.x;
  if (i >= n) return;
  float* row = out + (size_t)i * 40;
  float v[40];
  float m = -3.0e38f;
  #pragma unroll
  for (int c = 0; c < 40; ++c) { v[c] = row[c]; m = fmaxf(m, v[c]); }
  float s = 0.0f;
  #pragma unroll
  for (int c = 0; c < 40; ++c) s += __expf(v[c] - m);
  float ls = m + __logf(s);
  #pragma unroll
  for (int c = 0; c < 40; ++c) row[c] = v[c] - ls;
}

// ---------------------------------------------------------------------------

extern "C" void kernel_launch(void* const* d_in, const int* in_sizes, int n_in,
                              void* d_out, int out_size, void* d_ws, size_t ws_size,
                              hipStream_t stream) {
  const float* x     = (const float*)d_in[0];
  const int*   ei    = (const int*)d_in[1];
  const float* W1    = (const float*)d_in[2];
  const float* b1    = (const float*)d_in[3];
  const float* gamma = (const float*)d_in[4];
  const float* beta  = (const float*)d_in[5];
  const float* W2    = (const float*)d_in[6];
  const float* b2    = (const float*)d_in[7];
  float* out = (float*)d_out;

  const int N = in_sizes[0] / 256;   // 100000
  const int E = in_sizes[1] / 2;     // 1600000

  // workspace carve-out (256B-aligned slices)
  size_t off = 0;
  auto take = [&](size_t bytes) -> char* {
    char* r = (char*)d_ws + off;
    off += (bytes + 255) & ~(size_t)255;
    return r;
  };
  unsigned short* x_bf  = (unsigned short*)take((size_t)N * 256 * 2);
  unsigned short* w1t   = (unsigned short*)take((size_t)256 * 256 * 2);
  unsigned short* w2t   = (unsigned short*)take((size_t)64 * 256 * 2);
  float* h0             = (float*)take((size_t)N * 256 * 4);
  float* hagg           = (float*)take((size_t)N * 256 * 4);
  float* deg            = (float*)take((size_t)N * 4);
  float* dinv           = (float*)take((size_t)N * 4);
  float* bnsum          = (float*)take(256 * 4);
  float* bnsq           = (float*)take(256 * 4);
  float* bnscale        = (float*)take(256 * 4);
  float* bnshift        = (float*)take(256 * 4);
  // aliases over dead buffers (producer/consumer strictly ordered on stream):
  unsigned short* h_bf  = (unsigned short*)h0;   // h0 dead after k_edge_agg_h
  float* z              = (float*)x_bf;          // x_bf dead after GEMM1

  int total;

  total = N * 256;
  k_cast_bf16<<<(total + 255) / 256, 256, 0, stream>>>(x, x_bf, total);
  k_w1_tcast<<<256, 256, 0, stream>>>(W1, w1t);
  k_w2_tcast<<<64, 256, 0, stream>>>(W2, w2t);

  k_set_deg<<<(N + 255) / 256, 256, 0, stream>>>(deg, N);
  k_acc_deg<<<(E + 255) / 256, 256, 0, stream>>>(ei, deg, E);
  k_dinv<<<(N + 255) / 256, 256, 0, stream>>>(deg, dinv, N);

  { dim3 g(4, (N + 127) / 128);
    k_gemm_bf16<<<g, 256, 0, stream>>>(x_bf, w1t, h0, N, 256, 256); }

  total = N * 256;
  k_init_h<<<(total + 255) / 256, 256, 0, stream>>>(h0, dinv, b1, hagg, total);
  k_edge_agg_h<<<(E + 7) / 8, 256, 0, stream>>>(ei, h0, dinv, hagg, E);

  k_bn_zero<<<1, 256, 0, stream>>>(bnsum, bnsq);
  k_bn_partial<<<1024, 256, 0, stream>>>(hagg, bnsum, bnsq, N);
  k_bn_final<<<1, 256, 0, stream>>>(bnsum, bnsq, gamma, beta, bnscale, bnshift, N);
  k_bn_relu_cast<<<(total + 255) / 256, 256, 0, stream>>>(hagg, bnscale, bnshift, h_bf, total);

  { dim3 g(1, (N + 127) / 128);
    k_gemm_bf16<<<g, 256, 0, stream>>>(h_bf, w2t, z, N, 40, 40); }

  total = N * 40;
  k_init_out<<<(total + 255) / 256, 256, 0, stream>>>(z, dinv, b2, out, total);
  k_edge_agg_out<<<(E + 7) / 8, 256, 0, stream>>>(ei, z, dinv, out, E);
  k_logsoftmax<<<(N + 255) / 256, 256, 0, stream>>>(out, N);
}
